// CRF_Net_32744830664916
// MI455X (gfx1250) — matrix operand order repaired and checked by reading, not compiled, not run on hardware
//
#include <hip/hip_runtime.h>

#define DEVINL __device__ __forceinline__

typedef __bf16 v16bf __attribute__((ext_vector_type(16)));
typedef float  v8f   __attribute__((ext_vector_type(8)));
typedef unsigned short u16x8  __attribute__((ext_vector_type(8)));
typedef unsigned short u16x16 __attribute__((ext_vector_type(16)));

// problem dims
constexpr int B = 128, T = 1024, E = 32, H = 100, C = 20;
constexpr int G4 = 400;          // 4H gate width
constexpr int HP = 128;          // H padded to K-multiple of 32
constexpr int NT = 25;           // gate N-tiles of 16 (400/16)
constexpr int D1 = 200;          // 2H (layer-1 input / output feature width)
constexpr int D1P = 224;         // padded to 7*32
constexpr int NHP = 432;         // packed head rows: 20 f | 12 pad | 400 g
constexpr int NTH = 27;          // 432/16
constexpr int KCH = 7;           // 224/32
constexpr int MTB = 32;          // M tiles per block in heads GEMM

DEVINL unsigned short f2bf(float f) {
  union { float f; unsigned u; } v; v.f = f;
  unsigned r = v.u + 0x7FFFu + ((v.u >> 16) & 1u);   // round-to-nearest-even
  return (unsigned short)(r >> 16);
}
DEVINL float sigf(float x) { return 1.0f / (1.0f + __expf(-x)); }
DEVINL float tanhfast(float x) { float e = __expf(2.0f * x); return 1.0f - 2.0f / (e + 1.0f); }

// async 16-byte global -> LDS copy (ASYNCcnt); ldsPtr low 32 bits = LDS offset
DEVINL void asyncCopy16(void* ldsPtr, const void* gptr) {
  unsigned lds = (unsigned)(uintptr_t)ldsPtr;
  asm volatile("global_load_async_to_lds_b128 %0, %1, off" :: "v"(lds), "v"(gptr) : "memory");
}
DEVINL void asyncWait0() { asm volatile("s_wait_asynccnt 0x0" ::: "memory"); }
DEVINL void schedFence() {
#if __has_builtin(__builtin_amdgcn_sched_barrier)
  __builtin_amdgcn_sched_barrier(0);
#endif
}

// A-fragment (16-bit A 16x32 layout): lane<16 -> halves 0..7 = K0..7, 8..15 = K16..23
// lane>=16 -> halves 0..7 = K8..15, 8..15 = K24..31; row M = lane&15
DEVINL v16bf afrag(const unsigned short* row, int hi8) {
  u16x8 lo = *reinterpret_cast<const u16x8*>(row + hi8);
  u16x8 hi = *reinterpret_cast<const u16x8*>(row + 16 + hi8);
  u16x16 c = __builtin_shufflevector(lo, hi, 0,1,2,3,4,5,6,7,8,9,10,11,12,13,14,15);
  return __builtin_bit_cast(v16bf, c);
}
// B-fragment: pre-packed contiguous 16 halves per lane
DEVINL v16bf bfrag(const unsigned short* p) {
  u16x16 c = *reinterpret_cast<const u16x16*>(p);
  return __builtin_bit_cast(v16bf, c);
}

// ---------------- small prep kernels ----------------

__global__ void embedK(const int* __restrict__ tok, const float* __restrict__ emb,
                       unsigned short* __restrict__ X0) {
  int idx = blockIdx.x * blockDim.x + threadIdx.x;
  if (idx >= B * T * E) return;
  int k = idx % E; int bt = idx / E;
  X0[idx] = f2bf(emb[tok[bt] * E + k]);
}

// pack W[N][K] (row-major, op = W^T as B matrix) into per-lane WMMA B fragments
__global__ void packK(const float* __restrict__ W, unsigned short* __restrict__ out,
                      int N, int K, int ntiles, int kch) {
  int idx = blockIdx.x * blockDim.x + threadIdx.x;
  int total = ntiles * kch * 512;
  if (idx >= total) return;
  int j    = idx & 15;
  int lane = (idx >> 4) & 31;
  int kc   = (idx >> 9) % kch;
  int nt   = idx / (kch * 512);
  int n  = nt * 16 + (lane & 15);
  int kk = kc * 32 + ((lane < 16) ? 0 : 8) + (j & 7) + ((j & 8) ? 16 : 0);
  float v = (n < N && kk < K) ? W[(size_t)n * K + kk] : 0.0f;
  out[idx] = f2bf(v);
}

// packed head weight rows: [0,20) = fW, [20,32) = 0, [32,432) = gW
__global__ void concatWK(const float* __restrict__ fW, const float* __restrict__ gW,
                         float* __restrict__ Wc) {
  int idx = blockIdx.x * blockDim.x + threadIdx.x;
  if (idx >= NHP * D1) return;
  int n = idx / D1, k = idx % D1;
  float v = 0.0f;
  if (n < C) v = fW[n * D1 + k];
  else if (n >= 32) v = gW[(n - 32) * D1 + k];
  Wc[idx] = v;
}
__global__ void biasAddK(const float* __restrict__ a, const float* __restrict__ b,
                         float* __restrict__ o, int n) {
  int i = blockIdx.x * blockDim.x + threadIdx.x;
  if (i < n) o[i] = a[i] + b[i];
}
__global__ void concatBK(const float* __restrict__ fb, const float* __restrict__ gb,
                         float* __restrict__ o) {
  int i = blockIdx.x * blockDim.x + threadIdx.x;
  if (i >= NHP) return;
  float v = 0.0f;
  if (i < C) v = fb[i];
  else if (i >= 32) v = gb[i - 32];
  o[i] = v;
}

// ---------------- fused bidirectional LSTM layer ----------------
// 16 blocks: blockIdx[3] selects direction, low 3 bits the batch-tile.
// Weight B-fragments live in VGPRs for the whole sequence; x-tiles stream
// through double-buffered LDS via async global->LDS copies (ASYNCcnt).
template<int DIN, int DINP, int KCX>
__global__ __launch_bounds__(256) void lstm2K(
    const unsigned short* __restrict__ X,     // [B,T,DIN] bf16
    const unsigned short* __restrict__ wihA,  // fwd packed [NT][KCX][32][16]
    const unsigned short* __restrict__ whhA,  // fwd packed [NT][4][32][16]
    const float* __restrict__ biasA,
    const unsigned short* __restrict__ wihB,  // bwd
    const unsigned short* __restrict__ whhB,
    const float* __restrict__ biasB,
    const float* __restrict__ h0,             // [2L,B,H]
    const float* __restrict__ c0,
    int slotF, int slotR,
    unsigned short* __restrict__ Hcat) {      // [B,T,200] bf16
  __shared__ __align__(32) unsigned short sXb[2][16 * DINP];
  __shared__ __align__(32) unsigned short sH[16 * HP];
  __shared__ float sC[16 * H];
  __shared__ float sG[16 * G4];

  constexpr int SPR  = DIN / 8;       // b128 segments per row
  constexpr int NSEG = 16 * SPR;      // b128 segments per tile

  const int tid = threadIdx.x;
  const int dir = blockIdx.x >> 3;            // 0 = fwd, 1 = bwd
  const int bbase = (blockIdx.x & 7) * 16;
  const int rev = dir;
  const int dirOff = dir * H;
  const unsigned short* wih = dir ? wihB : wihA;
  const unsigned short* whh = dir ? whhB : whhA;
  const float* bias = dir ? biasB : biasA;
  const float* h0p = h0 + (size_t)(dir ? slotR : slotF) * B * H;
  const float* c0p = c0 + (size_t)(dir ? slotR : slotF) * B * H;

  // zero only pad columns (async copies own k < DIN)
  if constexpr (DINP > DIN) {
    constexpr int PADW = DINP - DIN;
    for (int e = tid; e < 2 * 16 * PADW; e += 256) {
      int kp = e % PADW, bm = e / PADW;
      int m = bm & 15, buf = bm >> 4;
      sXb[buf][m * DINP + DIN + kp] = 0;
    }
  }
  for (int e = tid; e < 16 * HP; e += 256) {
    int m = e / HP, j = e % HP;
    sH[e] = (j < H) ? f2bf(h0p[(size_t)(bbase + m) * H + j]) : (unsigned short)0;
  }
  for (int e = tid; e < 16 * H; e += 256) {
    int m = e / H, j = e % H;
    sC[e] = c0p[(size_t)(bbase + m) * H + j];
  }

  const int lane = tid & 31;
  const int wave = tid >> 5;
  const int hi8  = (lane & 16) ? 8 : 0;
  const int mrow = lane & 15;
  const int moff = (lane & 16) ? 8 : 0;

  // preload all weight B-fragments for this wave's N-tiles into registers
  v16bf wihR[4 * KCX];
  v16bf whhR[4 * 4];
  float bnq[4];
  #pragma unroll
  for (int q = 0; q < 4; ++q) {
    const int nt  = wave + 8 * q;
    const int ntc = (nt < NT) ? nt : 0;
    bnq[q] = bias[ntc * 16 + mrow];
    #pragma unroll
    for (int kc = 0; kc < KCX; ++kc)
      wihR[q * KCX + kc] = bfrag(wih + ((size_t)(ntc * KCX + kc) * 32 + lane) * 16);
    #pragma unroll
    for (int kc = 0; kc < 4; ++kc)
      whhR[q * 4 + kc] = bfrag(whh + ((size_t)(ntc * 4 + kc) * 32 + lane) * 16);
  }

  auto issueX = [&](int tt, int buf) {
    for (int e = tid; e < NSEG; e += 256) {
      int m = e / SPR, seg = e % SPR;
      asyncCopy16(&sXb[buf][m * DINP + seg * 8],
                  X + ((size_t)(bbase + m) * T + tt) * DIN + seg * 8);
    }
  };
  issueX(rev ? (T - 1) : 0, 0);
  __syncthreads();

  for (int step = 0; step < T; ++step) {
    const int cur = step & 1;
    const int tt = rev ? (T - 1 - step) : step;
    asyncWait0();          // own async copies done ...
    __syncthreads();       // ... and everyone else's; also orders prev elementwise
    if (step + 1 < T) issueX(rev ? (T - 2 - step) : (step + 1), cur ^ 1);

    // load all A-fragments, then the full WMMA burst behind one dscnt wait
    v16bf aX[KCX];
    v16bf aH[4];
    #pragma unroll
    for (int kc = 0; kc < KCX; ++kc) aX[kc] = afrag(&sXb[cur][mrow * DINP + kc * 32], hi8);
    #pragma unroll
    for (int kc = 0; kc < 4; ++kc)   aH[kc] = afrag(&sH[mrow * HP + kc * 32], hi8);
    schedFence();

    #pragma unroll
    for (int q = 0; q < 4; ++q) {
      const int nt = wave + 8 * q;          // wave-uniform
      if (nt < NT) {
        const int n = nt * 16 + mrow;
        v8f acc;
        #pragma unroll
        for (int r = 0; r < 8; ++r) acc[r] = bnq[q];
        #pragma unroll
        for (int kc = 0; kc < KCX; ++kc)
          acc = __builtin_amdgcn_wmma_f32_16x16x32_bf16(false, aX[kc], false, wihR[q * KCX + kc],
                                                        (short)0, acc, false, false);
        #pragma unroll
        for (int kc = 0; kc < 4; ++kc)
          acc = __builtin_amdgcn_wmma_f32_16x16x32_bf16(false, aH[kc], false, whhR[q * 4 + kc],
                                                        (short)0, acc, false, false);
        #pragma unroll
        for (int r = 0; r < 8; ++r) sG[(r + moff) * G4 + n] = acc[r];
      }
    }
    __syncthreads();

    for (int e = tid; e < 16 * H; e += 256) {
      int m = e / H, j = e % H;
      float gi = sG[m * G4 + j];
      float gf = sG[m * G4 + 100 + j];
      float gg = sG[m * G4 + 200 + j];
      float go = sG[m * G4 + 300 + j];
      float c = sigf(gf) * sC[e] + sigf(gi) * tanhfast(gg);
      float h = sigf(go) * tanhfast(c);
      sC[e] = c;
      unsigned short hbv = f2bf(h);
      sH[m * HP + j] = hbv;
      Hcat[((size_t)(bbase + m) * T + tt) * D1 + dirOff + j] = hbv;
    }
    // no trailing barrier: next iteration's top barrier orders these writes
  }
}

// ---------------- heads GEMM: [B*T,200] @ [432,200]^T + bias ----------------
// Weight-stationary waves + double-buffered async A-tiles.
__global__ __launch_bounds__(256) void headsK(
    const unsigned short* __restrict__ Hc,   // [B,T,200] bf16
    const unsigned short* __restrict__ wf,   // packed [NTH][KCH][32][16]
    const float* __restrict__ cb,            // [432]
    const int* __restrict__ seqlen,
    float* __restrict__ fout,                // [B,T,20]
    float* __restrict__ gout) {              // [B,T,400]
  __shared__ __align__(32) unsigned short sA2[2][16 * D1P];
  constexpr int SPRH  = D1 / 8;     // 25 b128 segments per row
  constexpr int NSEGH = 16 * SPRH;  // 400 per tile
  const int tid = threadIdx.x;
  const int lane = tid & 31, wave = tid >> 5;
  const int hi8 = (lane & 16) ? 8 : 0, mrow = lane & 15, moff = (lane & 16) ? 8 : 0;

  // zero pad columns of both buffers
  {
    constexpr int PADW = D1P - D1;
    for (int e = tid; e < 2 * 16 * PADW; e += 256) {
      int kp = e % PADW, bm = e / PADW;
      int m = bm & 15, buf = bm >> 4;
      sA2[buf][m * D1P + D1 + kp] = 0;
    }
  }

  const int nt  = blockIdx.y * 8 + wave;     // wave-uniform
  const bool act = (nt < NTH);
  const int ntc = act ? nt : 0;
  const int n   = ntc * 16 + mrow;
  const float bn = cb[n];

  v16bf bR[KCH];
  #pragma unroll
  for (int kc = 0; kc < KCH; ++kc)
    bR[kc] = bfrag(wf + ((size_t)(ntc * KCH + kc) * 32 + lane) * 16);

  auto issueA = [&](int mt, int buf) {
    const size_t r0 = ((size_t)blockIdx.x * MTB + mt) * 16;
    for (int e = tid; e < NSEGH; e += 256) {
      int m = e / SPRH, seg = e % SPRH;
      asyncCopy16(&sA2[buf][m * D1P + seg * 8], Hc + (r0 + m) * D1 + seg * 8);
    }
  };
  issueA(0, 0);

  for (int mt = 0; mt < MTB; ++mt) {
    const int cur = mt & 1;
    const size_t row0 = ((size_t)blockIdx.x * MTB + mt) * 16;
    asyncWait0();
    __syncthreads();
    // pad_packed_sequence mask fix-up (no-op when seq_lengths == T)
    if (tid < 16) {
      size_t row = row0 + tid;
      int t = (int)(row & (size_t)(T - 1));
      int b = (int)(row >> 10);
      if (t >= seqlen[b])
        for (int k = 0; k < D1; ++k) sA2[cur][tid * D1P + k] = 0;
    }
    __syncthreads();
    if (mt + 1 < MTB) issueA(mt + 1, cur ^ 1);

    if (act) {
      v16bf aR[KCH];
      #pragma unroll
      for (int kc = 0; kc < KCH; ++kc) aR[kc] = afrag(&sA2[cur][mrow * D1P + kc * 32], hi8);
      schedFence();
      v8f acc;
      #pragma unroll
      for (int r = 0; r < 8; ++r) acc[r] = bn;
      #pragma unroll
      for (int kc = 0; kc < KCH; ++kc)
        acc = __builtin_amdgcn_wmma_f32_16x16x32_bf16(false, aR[kc], false, bR[kc],
                                                      (short)0, acc, false, false);
      if (nt < 2) {                         // f tile group (wave-uniform)
        if (n < C) {
          #pragma unroll
          for (int r = 0; r < 8; ++r) fout[(row0 + r + moff) * C + n] = acc[r];
        }
      } else {                              // g tile group: always in-range
        const int ng = n - 32;
        #pragma unroll
        for (int r = 0; r < 8; ++r) gout[(row0 + r + moff) * (C * C) + ng] = acc[r];
      }
    }
  }
}

// ---------------- fused CRF forward+backward ----------------
// grid (B, 2): y==0 -> alpha scan, y==1 -> beta scan. One wave per block;
// next step's g/f blocks prefetched via async copies into LDS.
__global__ void crfK(const float* __restrict__ f, const float* __restrict__ g,
                     float* __restrict__ al, float* __restrict__ be) {
  const int b = blockIdx.x, j = threadIdx.x;
  __shared__ float sS[C];
  __shared__ __align__(16) float sG2[2][C * C];
  __shared__ __align__(16) float sF2[2][C];
  auto issue = [&](int ts, int buf) {
    const float* gp = g + ((size_t)b * T + ts) * (C * C);
    const float* fp = f + ((size_t)b * T + ts) * C;
    #pragma unroll
    for (int q = 0; q < 4; ++q) {
      int e = j + 32 * q;
      if (e < 100) asyncCopy16(&sG2[buf][e * 4], gp + e * 4);
    }
    if (j < 5) asyncCopy16(&sF2[buf][j * 4], fp + j * 4);
  };

  if (blockIdx.y == 0) {
    // alpha[0]=f[0]; alpha[t]=f[t]+LSE_i(alpha[t-1,i]+g[t,i,:])
    if (j < C) {
      float a0 = f[((size_t)b * T) * C + j];
      sS[j] = a0;
      al[((size_t)b * T) * C + j] = a0;
    }
    issue(1, 0);
    __syncthreads();
    for (int t = 1; t < T; ++t) {
      const int cur = (t - 1) & 1;
      asyncWait0();
      __syncthreads();
      if (t + 1 < T) issue(t + 1, cur ^ 1);
      float anew = 0.0f;
      if (j < C) {
        float tv[C]; float mx = -3.4e38f;
        #pragma unroll
        for (int i = 0; i < C; ++i) { tv[i] = sS[i] + sG2[cur][i * C + j]; mx = fmaxf(mx, tv[i]); }
        float s = 0.0f;
        #pragma unroll
        for (int i = 0; i < C; ++i) s += __expf(tv[i] - mx);
        anew = sF2[cur][j] + mx + __logf(s);
      }
      __syncthreads();
      if (j < C) { sS[j] = anew; al[((size_t)b * T + t) * C + j] = anew; }
      __syncthreads();
    }
  } else {
    // beta[T-1]=0; beta[t]=LSE_k(g[t+1,:,k]+f[t+1,k]+beta[t+1,k])
    if (j < C) { sS[j] = 0.0f; be[((size_t)b * T + (T - 1)) * C + j] = 0.0f; }
    issue(T - 1, 0);
    __syncthreads();
    for (int t = T - 2; t >= 0; --t) {
      const int cur = ((T - 2) - t) & 1;
      asyncWait0();
      __syncthreads();
      if (t > 0) issue(t, cur ^ 1);
      float bn = 0.0f;
      if (j < C) {
        float tv[C]; float mx = -3.4e38f;
        #pragma unroll
        for (int jj = 0; jj < C; ++jj) { tv[jj] = sG2[cur][j * C + jj] + sF2[cur][jj] + sS[jj]; mx = fmaxf(mx, tv[jj]); }
        float s = 0.0f;
        #pragma unroll
        for (int jj = 0; jj < C; ++jj) s += __expf(tv[jj] - mx);
        bn = mx + __logf(s);
      }
      __syncthreads();
      if (j < C) { sS[j] = bn; be[((size_t)b * T + t) * C + j] = bn; }
      __syncthreads();
    }
  }
}

__global__ void logzK(const float* __restrict__ al, float* __restrict__ lz) {
  int b = threadIdx.x;
  if (b >= B) return;
  const float* a = al + ((size_t)b * T + (T - 1)) * C;
  float mx = -3.4e38f;
  #pragma unroll
  for (int j = 0; j < C; ++j) mx = fmaxf(mx, a[j]);
  float s = 0.0f;
  #pragma unroll
  for (int j = 0; j < C; ++j) s += __expf(a[j] - mx);
  lz[b] = mx + __logf(s);
}

__global__ void margK(const float* __restrict__ al, const float* __restrict__ be,
                      const float* __restrict__ lz, float* __restrict__ out) {
  size_t idx = (size_t)blockIdx.x * blockDim.x + threadIdx.x;
  if (idx >= (size_t)B * T * C) return;
  int b = (int)(idx / ((size_t)T * C));
  out[idx] = __expf(al[idx] + be[idx] - lz[b]);
}

// ---------------- launch ----------------
extern "C" void kernel_launch(void* const* d_in, const int* in_sizes, int n_in,
                              void* d_out, int out_size, void* d_ws, size_t ws_size,
                              hipStream_t stream) {
  (void)in_sizes; (void)n_in; (void)out_size; (void)ws_size;
  const int*   tokens = (const int*)d_in[0];
  const int*   seqlen = (const int*)d_in[1];
  const float* emb    = (const float*)d_in[2];
  const float* Wih[4] = {(const float*)d_in[3], (const float*)d_in[7], (const float*)d_in[11], (const float*)d_in[15]};
  const float* Whh[4] = {(const float*)d_in[4], (const float*)d_in[8], (const float*)d_in[12], (const float*)d_in[16]};
  const float* bih[4] = {(const float*)d_in[5], (const float*)d_in[9], (const float*)d_in[13], (const float*)d_in[17]};
  const float* bhh[4] = {(const float*)d_in[6], (const float*)d_in[10], (const float*)d_in[14], (const float*)d_in[18]};
  const float* fW = (const float*)d_in[19];
  const float* fb = (const float*)d_in[20];
  const float* gW = (const float*)d_in[21];
  const float* gb = (const float*)d_in[22];
  const float* h0 = (const float*)d_in[23];
  const float* c0 = (const float*)d_in[24];
  float* out = (float*)d_out;

  char* ws = (char*)d_ws; size_t off = 0;
  auto alloc = [&](size_t bytes) -> char* {
    char* p = ws + off;
    off = (off + bytes + 255) & ~(size_t)255;
    return p;
  };
  unsigned short* X0  = (unsigned short*)alloc((size_t)B * T * E * 2);
  unsigned short* Hc0 = (unsigned short*)alloc((size_t)B * T * D1 * 2);
  unsigned short* Hc1 = (unsigned short*)alloc((size_t)B * T * D1 * 2);
  unsigned short* fWih[4]; unsigned short* fWhh[4]; float* cbias[4];
  const int kcx[4] = {1, 1, 7, 7};
  for (int d = 0; d < 4; ++d) {
    fWih[d]  = (unsigned short*)alloc((size_t)NT * kcx[d] * 512 * 2);
    fWhh[d]  = (unsigned short*)alloc((size_t)NT * 4 * 512 * 2);
    cbias[d] = (float*)alloc(G4 * 4);
  }
  unsigned short* fHead = (unsigned short*)alloc((size_t)NTH * KCH * 512 * 2);
  float* Wc = (float*)alloc((size_t)NHP * D1 * 4);
  float* hb = (float*)alloc(NHP * 4);
  float* lz = (float*)alloc(B * 4);

  const size_t BTC = (size_t)B * T * C, BTCC = (size_t)B * T * C * C;
  float* oMarg = out;
  float* oF = out + BTC;
  float* oG = out + 2 * BTC;
  float* oA = out + 2 * BTC + BTCC;
  float* oB = out + 3 * BTC + BTCC;

  hipLaunchKernelGGL(embedK, dim3((B * T * E + 255) / 256), dim3(256), 0, stream, tokens, emb, X0);
  for (int d = 0; d < 4; ++d) {
    int Kin = (d < 2) ? E : D1;
    hipLaunchKernelGGL(packK, dim3((NT * kcx[d] * 512 + 255) / 256), dim3(256), 0, stream,
                       Wih[d], fWih[d], G4, Kin, NT, kcx[d]);
    hipLaunchKernelGGL(packK, dim3((NT * 4 * 512 + 255) / 256), dim3(256), 0, stream,
                       Whh[d], fWhh[d], G4, H, NT, 4);
    hipLaunchKernelGGL(biasAddK, dim3(2), dim3(256), 0, stream, bih[d], bhh[d], cbias[d], G4);
  }
  hipLaunchKernelGGL(concatWK, dim3((NHP * D1 + 255) / 256), dim3(256), 0, stream, fW, gW, Wc);
  hipLaunchKernelGGL(packK, dim3((NTH * KCH * 512 + 255) / 256), dim3(256), 0, stream,
                     Wc, fHead, NHP, D1, NTH, KCH);
  hipLaunchKernelGGL(concatBK, dim3(2), dim3(256), 0, stream, fb, gb, hb);

  // layer 0 (both directions in one launch)
  hipLaunchKernelGGL((lstm2K<E, 32, 1>), dim3(16), dim3(256), 0, stream,
                     X0, fWih[0], fWhh[0], cbias[0], fWih[1], fWhh[1], cbias[1],
                     h0, c0, 0, 1, Hc0);
  // layer 1 (both directions in one launch)
  hipLaunchKernelGGL((lstm2K<D1, D1P, 7>), dim3(16), dim3(256), 0, stream,
                     Hc0, fWih[2], fWhh[2], cbias[2], fWih[3], fWhh[3], cbias[3],
                     h0, c0, 2, 3, Hc1);
  // emissions + transitions
  hipLaunchKernelGGL(headsK, dim3(B * T / 16 / MTB, 4), dim3(256), 0, stream,
                     Hc1, fHead, hb, seqlen, oF, oG);
  // CRF forward + backward (fused)
  hipLaunchKernelGGL(crfK, dim3(B, 2), dim3(32), 0, stream, oF, oG, oA, oB);
  hipLaunchKernelGGL(logzK, dim3(1), dim3(128), 0, stream, oA, lz);
  hipLaunchKernelGGL(margK, dim3((unsigned)((BTC + 255) / 256)), dim3(256), 0, stream, oA, oB, lz, oMarg);
}
